// ContrastiveLoss_50113678410454
// MI455X (gfx1250) — matrix operand order repaired and checked by reading, not compiled, not run on hardware
//
#include <hip/hip_runtime.h>
#include <math.h>

#define NB 128   // batch
#define NR 36    // regions
#define RP 48    // regions padded to WMMA multiple
#define NW 64    // words
#define ND 256   // feature dim
#define MARGIN_V 0.2f
#define LSM 9.0f
#define LLSE 6.0f
#define EPSV 1e-8f

#if defined(__has_builtin)
#  if __has_builtin(__builtin_amdgcn_global_load_async_to_lds_b128)
#    define HAVE_ASYNC_LDS 1
#  endif
#  if __has_builtin(__builtin_amdgcn_s_wait_asynccnt)
#    define HAVE_WAIT_ASYNC 1
#  endif
#endif
#ifndef HAVE_ASYNC_LDS
#  define HAVE_ASYNC_LDS 0
#endif
#ifndef HAVE_WAIT_ASYNC
#  define HAVE_WAIT_ASYNC 0
#endif

#if HAVE_WAIT_ASYNC
#  define WAIT_ASYNC(n) __builtin_amdgcn_s_wait_asynccnt(n)
#else
#  define WAIT_ASYNC(n) asm volatile("s_wait_asynccnt %0" ::"n"(n) : "memory")
#endif

typedef __attribute__((ext_vector_type(16))) __bf16 v16bf;
typedef __attribute__((ext_vector_type(8)))  float  v8f;
typedef __attribute__((ext_vector_type(4)))  int    v4i;
typedef __attribute__((address_space(1))) v4i gv4i;   // global int4 (prints as __device__)
typedef __attribute__((address_space(3))) v4i lv4i;   // LDS int4

union AFrag { v16bf v; uint4 q[2]; };
union F8    { v8f v; float f[8]; };

__device__ __forceinline__ unsigned short f2bf(float f) {
  unsigned int u = __float_as_uint(f);
  u += 0x7FFFu + ((u >> 16) & 1u);            // round-to-nearest-even
  return (unsigned short)(u >> 16);
}
__device__ __forceinline__ float bf2f(unsigned short h) {
  return __uint_as_float(((unsigned int)h) << 16);
}

// ---------- prep: im (f32 -> bf16, rows padded 36->48 with zeros) + ||im|| ----------
__global__ __launch_bounds__(64) void k_prep_im(const float* __restrict__ im,
                                                unsigned short* __restrict__ im_bf,
                                                float* __restrict__ n_im) {
  const int b = blockIdx.x;
  const int i = b / RP, r = b % RP;
  const int t = threadIdx.x;
  unsigned short* orow = im_bf + (size_t)(i * RP + r) * ND;
  __shared__ float red[64];
  float ss = 0.f;
  const int d0 = t * 4;
  if (r < NR) {
    const float4 x = *(const float4*)(im + (size_t)(i * NR + r) * ND + d0);
    orow[d0 + 0] = f2bf(x.x); orow[d0 + 1] = f2bf(x.y);
    orow[d0 + 2] = f2bf(x.z); orow[d0 + 3] = f2bf(x.w);
    ss = x.x * x.x + x.y * x.y + x.z * x.z + x.w * x.w;
  } else {
    orow[d0 + 0] = 0; orow[d0 + 1] = 0; orow[d0 + 2] = 0; orow[d0 + 3] = 0;
  }
  red[t] = ss;
  __syncthreads();
  for (int s = 32; s > 0; s >>= 1) { if (t < s) red[t] += red[t + s]; __syncthreads(); }
  if (t == 0 && r < NR) n_im[i * NR + r] = sqrtf(red[0]);
}

// ---------- prep: s (f32 -> bf16) ----------
__global__ __launch_bounds__(64) void k_prep_s(const float* __restrict__ s,
                                               unsigned short* __restrict__ s_bf) {
  const int b = blockIdx.x;                    // b = j*NW + w
  const int t = threadIdx.x;
  const int d0 = t * 4;
  const float4 x = *(const float4*)(s + (size_t)b * ND + d0);
  unsigned short* orow = s_bf + (size_t)b * ND;
  orow[d0 + 0] = f2bf(x.x); orow[d0 + 1] = f2bf(x.y);
  orow[d0 + 2] = f2bf(x.z); orow[d0 + 3] = f2bf(x.w);
}

// ---------- prep: per-caption Gram  G_j = s_j * s_j^T  (64x64, bf16) ----------
__global__ __launch_bounds__(256) void k_gram(const unsigned short* __restrict__ s_bf,
                                              unsigned short* __restrict__ g_bf) {
  const int j = blockIdx.x, t = threadIdx.x;
  __shared__ unsigned short sl[NW * ND];
  const uint4* src = (const uint4*)(s_bf + (size_t)j * NW * ND);
  uint4* dst = (uint4*)sl;
  for (int k = t; k < NW * ND / 8; k += 256) dst[k] = src[k];
  __syncthreads();
  for (int e = t; e < NW * NW; e += 256) {
    const int w = e >> 6, v = e & 63;
    const unsigned short* a = sl + w * ND;
    const unsigned short* b = sl + v * ND;
    float acc = 0.f;
    for (int d = 0; d < ND; ++d) acc += bf2f(a[d]) * bf2f(b[d]);
    g_bf[(size_t)j * NW * NW + e] = f2bf(acc);
  }
}

// ---------- main: one workgroup per (image i, caption j) pair ----------
__global__ __launch_bounds__(128) void k_pair(const unsigned short* __restrict__ s_bf,
                                              const unsigned short* __restrict__ im_bf,
                                              const unsigned short* __restrict__ g_bf,
                                              const float* __restrict__ n_im,
                                              const int* __restrict__ s_l,
                                              float* __restrict__ scores) {
  extern __shared__ unsigned char smem[];
  unsigned short* sS   = (unsigned short*)smem;            // s_j   : 64x256 bf16 (32KB)
  unsigned short* sI   = sS + NW * ND;                     // im_i  : 48x256 bf16 (24KB)
  unsigned short* sG   = sI + RP * ND;                     // G_j   : 64x64  bf16 ( 8KB)
  float*          er   = (float*)(sG + NW * NW);           // E_raw : 64x48  f32  (12KB)
  unsigned short* aT   = (unsigned short*)(er + NW * RP);  // a^T   : 48x64  bf16 ( 6KB)
  float*          rinv = (float*)(aT + RP * NW);           // 64
  float*          numr = rinv + NW;                        // 48
  float*          nwc2 = numr + RP;                        // 48
  float*          cmax = nwc2 + RP;                        // 48
  float*          ebuf = cmax + RP;                        // exp buf: 64x48 f32 (12KB)

  const int i = blockIdx.x, j = blockIdx.y;
  const int tid = threadIdx.x;
  const int wave = tid >> 5, lane = tid & 31;
  int len = s_l[j];
  if (len < 1) len = 1; if (len > NW) len = NW;

  // ---- stage operands into LDS; G is only needed after the softmax, so with
  //      async-to-LDS we leave its loads in flight across GEMM1 + softmax ----
  {
    const uint4* gs = (const uint4*)(s_bf + (size_t)j * NW * ND);
    const uint4* gi = (const uint4*)(im_bf + (size_t)i * RP * ND);
    const uint4* gg = (const uint4*)(g_bf + (size_t)j * NW * NW);
#if HAVE_ASYNC_LDS
    uint4* ls = (uint4*)sS;
    for (int k = tid; k < NW * ND / 8; k += 128)           // 16 instrs/wave
      __builtin_amdgcn_global_load_async_to_lds_b128((gv4i*)(gs + k), (lv4i*)(ls + k), 0, 0);
    uint4* li = (uint4*)sI;
    for (int k = tid; k < RP * ND / 8; k += 128)           // 12 instrs/wave
      __builtin_amdgcn_global_load_async_to_lds_b128((gv4i*)(gi + k), (lv4i*)(li + k), 0, 0);
    uint4* lg = (uint4*)sG;
    for (int k = tid; k < NW * NW / 8; k += 128)           // 4 instrs/wave (last)
      __builtin_amdgcn_global_load_async_to_lds_b128((gv4i*)(gg + k), (lv4i*)(lg + k), 0, 0);
    // async loads complete in order: <=4 outstanding => s and im have landed
    WAIT_ASYNC(4);
#else
    uint4* ls = (uint4*)sS;
    for (int k = tid; k < NW * ND / 8; k += 128) ls[k] = gs[k];
    uint4* li = (uint4*)sI;
    for (int k = tid; k < RP * ND / 8; k += 128) li[k] = gi[k];
    uint4* lg = (uint4*)sG;
    for (int k = tid; k < NW * NW / 8; k += 128) lg[k] = gg[k];
#endif
  }
  __syncthreads();

  // WMMA lane mapping (bf16 16x32 A / 32x16 B / f32 16x16 C); K is contiguous per lane half, not ISA table order
  const int mloc = lane & 15;
  const int kbA  = (lane < 16) ? 0 : 16;   // A: K group  {kbA..kbA+15}
  const int kbB  = (lane < 16) ? 0 : 16;   // B: K group  {kbB..kbB+15}
  const int moff = (lane < 16) ? 0 : 8;    // C: M offset per VGPR half
  const int m0   = wave * 16;              // each wave owns a 16-row band of E

  // ================= GEMM1: E_raw(64x48) = s_j(64x256) * im_i^T(256x48) ============
  v8f acc0 = {}, acc1 = {}, acc2 = {};
  for (int kk = 0; kk < 8; ++kk) {
    const int K0 = kk * 32;
    AFrag af;
    const unsigned short* ar = sS + (m0 + mloc) * ND + K0 + kbA;
    af.q[0] = *(const uint4*)(ar);
    af.q[1] = *(const uint4*)(ar + 8);
    const unsigned short* br = sI + mloc * ND + K0 + kbB;
    v16bf b0 = *(const v16bf*)(br);
    acc0 = __builtin_amdgcn_wmma_f32_16x16x32_bf16(false, af.v, false, b0, (short)0, acc0, false, false);
    v16bf b1 = *(const v16bf*)(br + 16 * ND);
    acc1 = __builtin_amdgcn_wmma_f32_16x16x32_bf16(false, af.v, false, b1, (short)0, acc1, false, false);
    v16bf b2 = *(const v16bf*)(br + 32 * ND);
    acc2 = __builtin_amdgcn_wmma_f32_16x16x32_bf16(false, af.v, false, b2, (short)0, acc2, false, false);
  }
  {
    F8 u0, u1, u2; u0.v = acc0; u1.v = acc1; u2.v = acc2;
    for (int v = 0; v < 8; ++v) {
      const int row = m0 + moff + v;
      er[row * RP + mloc]      = u0.f[v];
      er[row * RP + 16 + mloc] = u1.f[v];
      er[row * RP + 32 + mloc] = u2.f[v];
    }
  }
  __syncthreads();

  // ---- per-word inverse L2 norm over regions (padded cols are exact zeros) ----
  if (tid < NW) {
    float ss = 0.f;
    const float* e = er + tid * RP;
    for (int c = 0; c < RP; ++c) {
      const float x = e[c];
      const float l = (x >= 0.f) ? x : 0.1f * x;   // LeakyReLU(0.1)
      ss += l * l;
    }
    rinv[tid] = 1.f / (sqrtf(ss) + EPSV);
  }
  __syncthreads();

  // ---- masked softmax over words per region (exp spread across all 128 threads) ----
  if (tid < RP) {                                  // phase A: column maxima (no exp)
    const int r = tid;
    float mx = -1e30f;
    for (int w = 0; w < len; ++w) {
      const float x = er[w * RP + r];
      const float l = (x >= 0.f) ? x : 0.1f * x;
      mx = fmaxf(mx, LSM * l * rinv[w]);
    }
    cmax[r] = mx;
    nwc2[r] = 0.f;
  }
  __syncthreads();
  for (int idx = tid; idx < RP * NW; idx += 128) { // phase B: all exps exactly once
    const int r = idx >> 6, w = idx & 63;
    float e = 0.f;
    if (w < len) {
      const float x = er[w * RP + r];
      const float l = (x >= 0.f) ? x : 0.1f * x;
      e = __expf(LSM * l * rinv[w] - cmax[r]);
    }
    ebuf[w * RP + r] = e;
  }
  __syncthreads();
  if (tid < RP) {                                  // phase C: sums + a^T + num[r]
    const int r = tid;
    float ssum = 0.f;
    for (int w = 0; w < len; ++w) ssum += ebuf[w * RP + r];
    const float inv = 1.f / ssum;
    float nm = 0.f;
    for (int w = 0; w < NW; ++w) {
      const float a = ebuf[w * RP + r] * inv;
      aT[r * NW + w] = f2bf(a);                    // transposed => contiguous B-frags
      nm += a * er[w * RP + r];
    }
    numr[r] = nm;
  }
#if HAVE_ASYNC_LDS
  WAIT_ASYNC(0);                                   // G_j landed (issued last)
#endif
  __syncthreads();

  // ====== GEMM2: H(64x48) = G_j(64x64) * a(64x48);  ||wc_r||^2 = sum_w a*H ======
  v8f h0 = {}, h1 = {}, h2 = {};
  for (int kk = 0; kk < 2; ++kk) {
    const int K0 = kk * 32;
    AFrag af;
    const unsigned short* ar = sG + (m0 + mloc) * NW + K0 + kbA;
    af.q[0] = *(const uint4*)(ar);
    af.q[1] = *(const uint4*)(ar + 8);
    const unsigned short* br = aT + mloc * NW + K0 + kbB;
    v16bf b0 = *(const v16bf*)(br);
    h0 = __builtin_amdgcn_wmma_f32_16x16x32_bf16(false, af.v, false, b0, (short)0, h0, false, false);
    v16bf b1 = *(const v16bf*)(br + 16 * NW);
    h1 = __builtin_amdgcn_wmma_f32_16x16x32_bf16(false, af.v, false, b1, (short)0, h1, false, false);
    v16bf b2 = *(const v16bf*)(br + 32 * NW);
    h2 = __builtin_amdgcn_wmma_f32_16x16x32_bf16(false, af.v, false, b2, (short)0, h2, false, false);
  }
  {
    F8 u0, u1, u2; u0.v = h0; u1.v = h1; u2.v = h2;
    float p0 = 0.f, p1 = 0.f, p2 = 0.f;
    for (int v = 0; v < 8; ++v) {
      const int m = m0 + moff + v;
      p0 += bf2f(aT[(mloc)      * NW + m]) * u0.f[v];
      p1 += bf2f(aT[(16 + mloc) * NW + m]) * u1.f[v];
      p2 += bf2f(aT[(32 + mloc) * NW + m]) * u2.f[v];
    }
    atomicAdd(&nwc2[mloc], p0);
    atomicAdd(&nwc2[16 + mloc], p1);
    atomicAdd(&nwc2[32 + mloc], p2);
  }
  __syncthreads();

  // ---- cosine sim + LSE over regions -> score[i,j] ----
  if (tid < NR) {
    const float nw = sqrtf(fmaxf(nwc2[tid], 0.f));
    const float den = fmaxf(n_im[i * NR + tid] * nw, EPSV);
    numr[tid] = numr[tid] / den;                   // row_sim
  }
  __syncthreads();
  if (tid == 0) {
    float mx = -1e30f;
    for (int r = 0; r < NR; ++r) mx = fmaxf(mx, LLSE * numr[r]);
    float ssum = 0.f;
    for (int r = 0; r < NR; ++r) ssum += __expf(LLSE * numr[r] - mx);
    scores[i * NB + j] = (__logf(ssum) + mx) / LLSE;
  }
}

// ---------- hardest-negative hinge loss over 128x128 score matrix ----------
__global__ __launch_bounds__(128) void k_loss(const float* __restrict__ S,
                                              float* __restrict__ out) {
  __shared__ float red[128];
  const int t = threadIdx.x;
  const float dt = S[t * NB + t];
  float mrow = 0.f, mcol = 0.f;                    // relu + zeroed diag => >= 0
  for (int k = 0; k < NB; ++k) {
    if (k == t) continue;
    mrow = fmaxf(mrow, MARGIN_V + S[t * NB + k] - dt);  // cost_s row max
    mcol = fmaxf(mcol, MARGIN_V + S[k * NB + t] - dt);  // cost_im col max
  }
  red[t] = mrow + mcol;
  __syncthreads();
  for (int s = 64; s > 0; s >>= 1) { if (t < s) red[t] += red[t + s]; __syncthreads(); }
  if (t == 0) out[0] = red[0];
}

extern "C" void kernel_launch(void* const* d_in, const int* in_sizes, int n_in,
                              void* d_out, int out_size, void* d_ws, size_t ws_size,
                              hipStream_t stream) {
  (void)in_sizes; (void)n_in; (void)out_size; (void)ws_size;
  const float* im = (const float*)d_in[0];
  const float* s  = (const float*)d_in[1];
  const int*   sl = (const int*)d_in[2];
  float* out = (float*)d_out;

  unsigned char* ws = (unsigned char*)d_ws;
  size_t off = 0;
  unsigned short* s_bf  = (unsigned short*)(ws + off); off += (size_t)NB * NW * ND * 2;
  unsigned short* im_bf = (unsigned short*)(ws + off); off += (size_t)NB * RP * ND * 2;
  unsigned short* g_bf  = (unsigned short*)(ws + off); off += (size_t)NB * NW * NW * 2;
  float* n_im   = (float*)(ws + off); off += (size_t)NB * NR * 4;
  float* scores = (float*)(ws + off); off += (size_t)NB * NB * 4;

  k_prep_s <<<NB * NW, 64, 0, stream>>>(s, s_bf);
  k_prep_im<<<NB * RP, 64, 0, stream>>>(im, im_bf, n_im);
  k_gram   <<<NB, 256, 0, stream>>>(s_bf, g_bf);

  const size_t lds = (size_t)(NW * ND * 2 + RP * ND * 2 + NW * NW * 2 +
                              NW * RP * 4 + RP * NW * 2 +
                              NW * 4 + RP * 4 + RP * 4 + RP * 4 +
                              NW * RP * 4);
  dim3 grid(NB, NB);
  k_pair<<<grid, 128, lds, stream>>>(s_bf, im_bf, g_bf, n_im, sl, scores);
  k_loss<<<1, NB, 0, stream>>>(scores, out);
}
